// xDeepFM_3831110828056
// MI455X (gfx1250) — matrix operand order repaired and checked by reading, not compiled, not run on hardware
//
#include <hip/hip_runtime.h>
#include <hip/hip_bf16.h>

// ---------------------------------------------------------------------------
// xDeepFM forward for MI455X (gfx1250, wave32, WMMA bf16 16x16x32)
// ---------------------------------------------------------------------------

typedef __attribute__((ext_vector_type(16))) __bf16 v16bf;
typedef __attribute__((ext_vector_type(8)))  float  v8f;

#define BATCH   2048
#define NF      39
#define ED      16
#define CIN_SZ  200      // each CIN layer output rows
#define DNN_SZ  400

__device__ __forceinline__ unsigned short f2bf(float f) {
    unsigned int u = __float_as_uint(f);
    unsigned int r = (u + 0x7FFFu + ((u >> 16) & 1u)) >> 16;   // RNE
    return (unsigned short)r;
}
__device__ __forceinline__ float bf2f(unsigned short h) {
    return __uint_as_float(((unsigned int)h) << 16);
}

// ---------------------------------------------------------------------------
// 1) Embedding gather: inputs_bf[b, f, d] = bf16(emb[id]*val); bias row sums.
// ---------------------------------------------------------------------------
__global__ __launch_bounds__(64) void gather_kernel(
    const int* __restrict__ ids, const float* __restrict__ vals,
    const float* __restrict__ emb, const float* __restrict__ bias_emb,
    unsigned short* __restrict__ inputs_bf, float* __restrict__ biassum)
{
    int b = blockIdx.x;
    int f = threadIdx.x;
    __shared__ float sb[NF];
    if (f < NF) {
        int   id = ids[(size_t)b * NF + f];
        float v  = vals[(size_t)b * NF + f];
        const float* er = emb + (size_t)id * ED;
        unsigned short* op = inputs_bf + ((size_t)b * NF + f) * ED;
#pragma unroll
        for (int d = 0; d < ED; ++d) op[d] = f2bf(er[d] * v);
        sb[f] = bias_emb[id] * v;
    }
    __syncthreads();
    if (f == 0) {
        float s = 0.f;
        for (int i = 0; i < NF; ++i) s += sb[i];
        biassum[b] = s;
    }
}

// ---------------------------------------------------------------------------
// 2a) Pack CIN weight (200 x Kreal, f32) into WMMA A-fragment order (bf16).
//     tile = kt*13 + mt ; per tile: 32 lanes x 16 bf16 (A 16x32 layout).
// ---------------------------------------------------------------------------
__global__ __launch_bounds__(256) void pack_cinw(
    const float* __restrict__ W, unsigned short* __restrict__ wpack,
    int Kreal, int KT)
{
    int t = blockIdx.x * blockDim.x + threadIdx.x;
    int ntile = KT * 13;
    if (t >= ntile * 32) return;
    int tile = t >> 5, lane = t & 31;
    int kt = tile / 13, mt = tile % 13;
    int row  = mt * 16 + (lane & 15);
    int half = lane >> 4;
    unsigned int* dst = (unsigned int*)wpack + (size_t)tile * 256 + lane * 8;
#pragma unroll
    for (int r = 0; r < 8; ++r) {
        int k0 = kt * 32 + half * 16 + 2 * r;
        float a0 = (row < CIN_SZ && k0     < Kreal) ? W[(size_t)row * Kreal + k0]     : 0.f;
        float a1 = (row < CIN_SZ && k0 + 1 < Kreal) ? W[(size_t)row * Kreal + k0 + 1] : 0.f;
        dst[r] = (unsigned int)f2bf(a0) | ((unsigned int)f2bf(a1) << 16);
    }
}

// ---------------------------------------------------------------------------
// 2b) Pack DNN weight (N x Kreal, f32) into WMMA B-fragment order (bf16).
//     B 32x16: lane = N col, half selects K group, VGPR r -> K = half*16+2r.
//     tile = kt*NT + nt.
// ---------------------------------------------------------------------------
__global__ __launch_bounds__(256) void pack_dnnw(
    const float* __restrict__ W, unsigned short* __restrict__ wpack,
    int Kreal, int KT, int Nreal, int NT)
{
    int t = blockIdx.x * blockDim.x + threadIdx.x;
    int ntile = KT * NT;
    if (t >= ntile * 32) return;
    int tile = t >> 5, lane = t & 31;
    int kt = tile / NT, nt = tile % NT;
    int o    = nt * 16 + (lane & 15);
    int half = lane >> 4;
    unsigned int* dst = (unsigned int*)wpack + (size_t)tile * 256 + lane * 8;
#pragma unroll
    for (int r = 0; r < 8; ++r) {
        int k0 = kt * 32 + half * 16 + 2 * r;
        float a0 = (o < Nreal && k0     < Kreal) ? W[(size_t)o * Kreal + k0]     : 0.f;
        float a1 = (o < Nreal && k0 + 1 < Kreal) ? W[(size_t)o * Kreal + k0 + 1] : 0.f;
        dst[r] = (unsigned int)f2bf(a0) | ((unsigned int)f2bf(a1) << 16);
    }
}

// ---------------------------------------------------------------------------
// 3) CIN layer. One wave per sample; 4 waves (128 threads) per block.
//    out[o,d] = relu( sum_c W[o,c] * hprev[c/39,d]*inputs[c%39,d] + bias[o] )
//    M = 200 (13 tiles), N = 16 (=EMB_DIM), K = Kreal (KT tiles of 32).
//    LDS rows are zero-padded one row past the end so the padded K range
//    (up to KT*32-1) contributes exact zeros -> no predication in hot loop.
//    A fragments stream with a depth-2 prefetch pipeline.
// ---------------------------------------------------------------------------
#define INS_ROWS 40      // 39 + 1 zero row
#define HPV_ROWS 101     // 100 + 1 zero row

__global__ __launch_bounds__(128) void cin_kernel(
    const unsigned short* __restrict__ inputs_bf,   // [B,39,16] bf16
    const unsigned short* __restrict__ hprev,       // [B,100,16] bf16 or null
    const unsigned short* __restrict__ wpack,       // packed A fragments
    const float* __restrict__ bias,                 // [200]
    unsigned short* __restrict__ hnext,             // [B,100,16] bf16 or null
    float* __restrict__ cinvec,                     // [B,400]
    int Kreal, int KT, int dstart, int coff)
{
    const int wave = threadIdx.x >> 5;
    const int lane = threadIdx.x & 31;
    const int b = blockIdx.x * 4 + wave;

    __shared__ unsigned short sh[4][INS_ROWS * ED + HPV_ROWS * ED];
    unsigned short* ins = &sh[wave][0];
    unsigned short* hpv = ins + INS_ROWS * ED;

    {   // stage this wave's sample into LDS (dword copies) + zero pad rows
        unsigned int* s32 = (unsigned int*)ins;
        const unsigned int* g32 = (const unsigned int*)(inputs_bf + (size_t)b * 624);
        for (int i = lane; i < 312; i += 32) s32[i] = g32[i];
        if (lane < 8) s32[312 + lane] = 0u;              // row 39 = 0
        if (hprev) {
            unsigned int* h32 = (unsigned int*)hpv;
            const unsigned int* gh = (const unsigned int*)(hprev + (size_t)b * 1600);
            for (int i = lane; i < 800; i += 32) h32[i] = gh[i];
            if (lane < 8) h32[800 + lane] = 0u;          // row 100 = 0
        }
    }
    __syncthreads();
    const unsigned short* hs = hprev ? hpv : ins;   // layer0: hprev == inputs

    const int n    = lane & 15;
    const int half = lane >> 4;

    v8f acc[13];
#pragma unroll
    for (int m = 0; m < 13; ++m) acc[m] = (v8f){0.f,0.f,0.f,0.f,0.f,0.f,0.f,0.f};

    const v16bf* wp = (const v16bf*)wpack;     // flat tiles: [kt*13 + mt]
    const int T = KT * 13;
    // depth-2 A-fragment prefetch pipeline
    v16bf afA = wp[(size_t)0 * 32 + lane];
    v16bf afB = wp[(size_t)1 * 32 + lane];

    for (int kt = 0; kt < KT; ++kt) {
        // ---- build B fragment (x tile, 32x16) from LDS, no predication ----
        union { unsigned int u[8]; v16bf v; } bfrag;
        int c = kt * 32 + half * 16;
        int h = c / 39;            // one division per k-tile
        int m = c - h * 39;
#pragma unroll
        for (int r = 0; r < 8; ++r) {
            int m1 = m + 1, h1 = h;
            if (m1 == 39) { m1 = 0; h1 = h + 1; }
            unsigned int pa = (unsigned int)ins[m  * ED + n]
                            | ((unsigned int)ins[m1 * ED + n] << 16);
            unsigned int pb = (unsigned int)hs[h  * ED + n]
                            | ((unsigned int)hs[h1 * ED + n] << 16);
            unsigned int pk;
            asm("v_pk_mul_bf16 %0, %1, %2" : "=v"(pk) : "v"(pa), "v"(pb));
            bfrag.u[r] = pk;
            m += 2; if (m >= 39) { m -= 39; ++h; }
        }
        // ---- 13 WMMAs; prefetch A tiles two steps ahead ----
        int t0 = kt * 13;
#pragma unroll
        for (int mt = 0; mt < 13; ++mt) {
            int tn = t0 + mt + 2;
            if (tn > T - 1) tn = T - 1;
            v16bf afn = wp[(size_t)tn * 32 + lane];
            acc[mt] = __builtin_amdgcn_wmma_f32_16x16x32_bf16(
                false, afA, false, bfrag.v, (short)0, acc[mt], false, false);
            afA = afB; afB = afn;
        }
    }

    // Epilogue: bias + relu; row-sum over d (16 N-lanes); emit hnext / cinvec.
#pragma unroll
    for (int mt = 0; mt < 13; ++mt) {
#pragma unroll
        for (int r = 0; r < 8; ++r) {
            int row = mt * 16 + half * 8 + r;
            float v = acc[mt][r];
            if (row < CIN_SZ) {
                v += bias[row];
                v = v > 0.f ? v : 0.f;
            } else v = 0.f;
            float s = v;
            s += __shfl_xor(s, 1);
            s += __shfl_xor(s, 2);
            s += __shfl_xor(s, 4);
            s += __shfl_xor(s, 8);
            if (row < CIN_SZ) {
                if (n == 0 && row >= dstart)
                    cinvec[(size_t)b * 400 + coff + (row - dstart)] = s;
                if (hnext && row < 100)
                    hnext[((size_t)b * 100 + row) * 16 + n] = f2bf(v);
            }
        }
    }
}

// ---------------------------------------------------------------------------
// 4) DNN GEMM: out[M=2048, N=400] = relu(A[M,K] * W^T + bias), bf16 WMMA.
//    A row-major bf16; B fragments pre-packed (depth-1 prefetch).
// ---------------------------------------------------------------------------
__global__ __launch_bounds__(128) void dnn_gemm(
    const unsigned short* __restrict__ A,       // [2048, Kreal] bf16
    const unsigned short* __restrict__ bpack,   // packed B fragments
    const float* __restrict__ bias,             // [400]
    unsigned short* __restrict__ out_bf,        // [2048,400] bf16 (or null)
    float* __restrict__ out_f,                  // [2048,400] f32  (or null)
    int Kreal, int KT, int NT)
{
    const int MT = BATCH / 16;                  // 128
    int wid = blockIdx.x * 4 + (threadIdx.x >> 5);
    if (wid >= MT * NT) return;
    int mt = wid / NT, nt = wid % NT;
    int lane = threadIdx.x & 31;
    int n = lane & 15, half = lane >> 4;
    int mrow = mt * 16 + n;                     // A layout: M across lanes

    v8f acc = (v8f){0.f,0.f,0.f,0.f,0.f,0.f,0.f,0.f};
    const unsigned int* arow = (const unsigned int*)(A + (size_t)mrow * Kreal);
    const v16bf* bp = (const v16bf*)bpack;
    v16bf bf = bp[(size_t)(0 * NT + nt) * 32 + lane];
    for (int kt = 0; kt < KT; ++kt) {
        int ktn = (kt + 1 < KT) ? kt + 1 : kt;
        v16bf bfn = bp[(size_t)(ktn * NT + nt) * 32 + lane];
        union { unsigned int u[8]; v16bf v; } afrag;
#pragma unroll
        for (int r = 0; r < 8; ++r) {
            int k0 = kt * 32 + half * 16 + 2 * r;
            afrag.u[r] = (k0 < Kreal) ? arow[k0 >> 1] : 0u;
        }
        acc = __builtin_amdgcn_wmma_f32_16x16x32_bf16(
            false, afrag.v, false, bf, (short)0, acc, false, false);
        bf = bfn;
    }
    int ocol = nt * 16 + n;
    float bi = bias[ocol];
#pragma unroll
    for (int r = 0; r < 8; ++r) {
        int brow = mt * 16 + half * 8 + r;
        float v = acc[r] + bi;
        v = v > 0.f ? v : 0.f;
        if (out_bf) out_bf[(size_t)brow * DNN_SZ + ocol] = f2bf(v);
        else        out_f [(size_t)brow * DNN_SZ + ocol] = v;
    }
}

// ---------------------------------------------------------------------------
// 5) Final: out[b] = dot(concat(cinvec, d1), fc_W) + fc_b + biassum[b]
// ---------------------------------------------------------------------------
__global__ __launch_bounds__(128) void final_kernel(
    const float* __restrict__ cinvec, const float* __restrict__ d1,
    const float* __restrict__ fcW, const float* __restrict__ fcb,
    const float* __restrict__ biassum, float* __restrict__ out)
{
    int b = blockIdx.x * 4 + (threadIdx.x >> 5);
    int lane = threadIdx.x & 31;
    float s = 0.f;
    for (int i = lane; i < 800; i += 32) {
        float x = (i < 400) ? cinvec[(size_t)b * 400 + i]
                            : d1[(size_t)b * 400 + (i - 400)];
        s += x * fcW[i];
    }
    s += __shfl_xor(s, 16);
    s += __shfl_xor(s, 8);
    s += __shfl_xor(s, 4);
    s += __shfl_xor(s, 2);
    s += __shfl_xor(s, 1);
    if (lane == 0) out[b] = s + fcb[0] + biassum[b];
}

// ---------------------------------------------------------------------------
extern "C" void kernel_launch(void* const* d_in, const int* in_sizes, int n_in,
                              void* d_out, int out_size, void* d_ws, size_t ws_size,
                              hipStream_t stream)
{
    const int*   feature_ids = (const int*)  d_in[0];
    const float* feature_vals= (const float*)d_in[1];
    const float* emb         = (const float*)d_in[2];
    const float* bias_emb    = (const float*)d_in[3];
    const float* cin_W0      = (const float*)d_in[4];
    const float* cin_b0      = (const float*)d_in[5];
    const float* cin_W1      = (const float*)d_in[6];
    const float* cin_b1      = (const float*)d_in[7];
    const float* cin_W2      = (const float*)d_in[8];
    const float* cin_b2      = (const float*)d_in[9];
    const float* dnn_W0      = (const float*)d_in[10];
    const float* dnn_b0      = (const float*)d_in[11];
    const float* dnn_W1      = (const float*)d_in[12];
    const float* dnn_b1      = (const float*)d_in[13];
    const float* fc_W        = (const float*)d_in[14];
    const float* fc_b        = (const float*)d_in[15];
    float* out = (float*)d_out;

    // ---- workspace carve-out (256B aligned) ----
    char* p = (char*)d_ws;
    auto alloc = [&](size_t bytes) -> char* {
        char* r = p; p += (bytes + 255) & ~(size_t)255; return r;
    };
    unsigned short* inputs_bf = (unsigned short*)alloc((size_t)BATCH * NF * ED * 2);
    float*          biassum   = (float*)         alloc((size_t)BATCH * 4);
    unsigned short* hA        = (unsigned short*)alloc((size_t)BATCH * 100 * ED * 2);
    unsigned short* hB        = (unsigned short*)alloc((size_t)BATCH * 100 * ED * 2);
    float*          cinvec    = (float*)         alloc((size_t)BATCH * 400 * 4);
    unsigned short* d0bf      = (unsigned short*)alloc((size_t)BATCH * 400 * 2);
    float*          d1f       = (float*)         alloc((size_t)BATCH * 400 * 4);
    unsigned short* wp0       = (unsigned short*)alloc((size_t)48  * 13 * 1024);
    unsigned short* wp1       = (unsigned short*)alloc((size_t)122 * 13 * 1024);
    unsigned short* wp2       = (unsigned short*)alloc((size_t)122 * 13 * 1024);
    unsigned short* w0p       = (unsigned short*)alloc((size_t)20  * 25 * 1024);
    unsigned short* w1p       = (unsigned short*)alloc((size_t)13  * 25 * 1024);
    (void)ws_size; (void)in_sizes; (void)n_in; (void)out_size;

    // 1) gather embeddings + bias sums
    gather_kernel<<<BATCH, 64, 0, stream>>>(feature_ids, feature_vals, emb,
                                            bias_emb, inputs_bf, biassum);

    // 2) pack weights into WMMA fragment layouts (bf16)
    {
        int t0 = 48 * 13 * 32, t1 = 122 * 13 * 32;
        pack_cinw<<<(t0 + 255) / 256, 256, 0, stream>>>(cin_W0, wp0, 1521, 48);
        pack_cinw<<<(t1 + 255) / 256, 256, 0, stream>>>(cin_W1, wp1, 3900, 122);
        pack_cinw<<<(t1 + 255) / 256, 256, 0, stream>>>(cin_W2, wp2, 3900, 122);
        int d0 = 20 * 25 * 32, d1 = 13 * 25 * 32;
        pack_dnnw<<<(d0 + 255) / 256, 256, 0, stream>>>(dnn_W0, w0p, 624, 20, 400, 25);
        pack_dnnw<<<(d1 + 255) / 256, 256, 0, stream>>>(dnn_W1, w1p, 400, 13, 400, 25);
    }

    // 3) CIN layers (launch order = global sync between layers)
    cin_kernel<<<BATCH / 4, 128, 0, stream>>>(inputs_bf, nullptr, wp0, cin_b0,
                                              hA, cinvec, 1521, 48, 100, 0);
    cin_kernel<<<BATCH / 4, 128, 0, stream>>>(inputs_bf, hA, wp1, cin_b1,
                                              hB, cinvec, 3900, 122, 100, 100);
    cin_kernel<<<BATCH / 4, 128, 0, stream>>>(inputs_bf, hB, wp2, cin_b2,
                                              nullptr, cinvec, 3900, 122, 0, 200);

    // 4) DNN
    {
        int wids = (BATCH / 16) * 25;   // 3200 waves
        dnn_gemm<<<wids / 4, 128, 0, stream>>>(inputs_bf, w0p, dnn_b0,
                                               d0bf, nullptr, 624, 20, 25);
        dnn_gemm<<<wids / 4, 128, 0, stream>>>(d0bf, w1p, dnn_b1,
                                               nullptr, d1f, 400, 13, 25);
    }

    // 5) final logit
    final_kernel<<<BATCH / 4, 128, 0, stream>>>(cinvec, d1f, fc_W, fc_b,
                                                biassum, out);
}
